// CausalWanSelfAttention_6975026889198
// MI455X (gfx1250) — matrix operand order repaired and checked by reading, not compiled, not run on hardware
//
#include <hip/hip_runtime.h>
#include <hip/hip_bf16.h>
#include <math.h>

#define F_FRAMES 9
#define H_GRID   16
#define W_GRID   28
#define S_TOK    (F_FRAMES * H_GRID * W_GRID)   /* 4032 */
#define HW_GRID  (H_GRID * W_GRID)              /* 448  */
#define DIM      1536
#define N_HEADS  12
#define HEAD_DIM 128
#define ROT      64
#define EPS_F    1e-6f
#define QK_SCALE 0.08838834764831845f           /* 1/sqrt(128) */

typedef __bf16 bf16_t;
typedef __attribute__((ext_vector_type(16))) __bf16 v16bf;
typedef __attribute__((ext_vector_type(8)))  __bf16 v8bf;
typedef __attribute__((ext_vector_type(8)))  float  v8f;

// ---------------------------------------------------------------------------
// f32 -> bf16 cast (grid-stride), used for the GEMM A operands (row-major)
// ---------------------------------------------------------------------------
__global__ void cast_f32_bf16(const float* __restrict__ in,
                              bf16_t* __restrict__ out, int n) {
  for (int i = blockIdx.x * blockDim.x + threadIdx.x; i < n;
       i += gridDim.x * blockDim.x) {
    out[i] = (bf16_t)in[i];
  }
}

// ---------------------------------------------------------------------------
// f32 -> bf16 cast + transpose: in is R x C (row-major), out is C x R.
// LDS-tiled 32x32 so both global streams are coalesced. Used to produce
// W^T (so B-fragments are contiguous) and V^T (so PV B-fragments are
// contiguous over keys).
// ---------------------------------------------------------------------------
__global__ void cast_transpose_f32_bf16(const float* __restrict__ in,
                                        bf16_t* __restrict__ out,
                                        int R, int C) {
  __shared__ float tile[32][33];
  const int c0 = blockIdx.x * 32;
  const int r0 = blockIdx.y * 32;
  for (int i = threadIdx.y; i < 32; i += 8)
    tile[i][threadIdx.x] = in[(size_t)(r0 + i) * C + c0 + threadIdx.x];
  __syncthreads();
  for (int i = threadIdx.y; i < 32; i += 8)
    out[(size_t)(c0 + i) * R + r0 + threadIdx.x] = (bf16_t)tile[threadIdx.x][i];
}

// ---------------------------------------------------------------------------
// GEMM: C[M,N] = A[M,K](bf16 row-major) * B[K,N] + bias[N], with B supplied
// TRANSPOSED (Bt is N x K, row-major). 128 threads = 4 waves; each wave owns
// a 16x64 output slab: one A-fragment feeds 4 WMMAs per 32-deep k-step.
// All fragments are contiguous b128 loads (k order used here, same for A and B):
//   A 16x32 : lane half 0 -> K{0..15},       half 1 -> K{16..31} (row fixed)
//   B 32x16 : lane half 0 -> K{0..15},       half 1 -> K{16..31} (col fixed)
// ---------------------------------------------------------------------------
__global__ void gemm_bf16_f32(const bf16_t* __restrict__ A,
                              const bf16_t* __restrict__ Bt,
                              const float* __restrict__ bias,
                              float* __restrict__ C,
                              int M, int N, int K) {
  const int lane = threadIdx.x & 31;
  const int wid  = threadIdx.x >> 5;
  const int r    = lane & 15;
  const int half = lane >> 4;

  const int row = blockIdx.y * 16 + r;
  const int n0  = (blockIdx.x * 4 + wid) * 64;

  const bf16_t* arow = A + (size_t)row * K;
  v8f acc[4];
#pragma unroll
  for (int t = 0; t < 4; ++t) acc[t] = (v8f){};

  for (int k0 = 0; k0 < K; k0 += 32) {
    __builtin_prefetch(arow + k0 + 512, 0, 1);     // global_prefetch_b8
    const v8bf alo = *(const v8bf*)(arow + k0 + half * 16);
    const v8bf ahi = *(const v8bf*)(arow + k0 + 8 + half * 16);
    v16bf a;
#pragma unroll
    for (int e = 0; e < 8; ++e) { a[e] = alo[e]; a[e + 8] = ahi[e]; }
#pragma unroll
    for (int t = 0; t < 4; ++t) {
      const v16bf b =
          *(const v16bf*)(Bt + (size_t)(n0 + t * 16 + r) * K + k0 + 16 * half);
      acc[t] = __builtin_amdgcn_wmma_f32_16x16x32_bf16(
          false, a, false, b, (short)0, acc[t], false, false);
    }
  }
#pragma unroll
  for (int t = 0; t < 4; ++t) {
    const int col = n0 + t * 16 + r;
#pragma unroll
    for (int i = 0; i < 8; ++i) {
      const int m = blockIdx.y * 16 + i + 8 * half;  // C/D: VGPR i -> M = i + 8*half
      C[(size_t)m * N + col] = acc[t][i] + bias[col];
    }
  }
}

// ---------------------------------------------------------------------------
// Fused RMSNorm (over DIM) + RoPE (per head, interleaved pairs) + bf16 cast.
// One 256-thread block per token. `mul` folds the attention 1/sqrt(d) into Q.
// ---------------------------------------------------------------------------
__global__ void rmsnorm_rope_cast(const float* __restrict__ in,
                                  const float* __restrict__ g,
                                  const float* __restrict__ freqs,
                                  bf16_t* __restrict__ out, float mul) {
  const int s   = blockIdx.x;
  const int tid = threadIdx.x;

  float ss = 0.f;
  for (int j = tid; j < DIM; j += 256) {
    float v = in[(size_t)s * DIM + j];
    ss += v * v;
  }
#pragma unroll
  for (int off = 16; off >= 1; off >>= 1) ss += __shfl_xor(ss, off, 32);
  __shared__ float part[8];
  if ((tid & 31) == 0) part[tid >> 5] = ss;
  __syncthreads();
  float tot = 0.f;
#pragma unroll
  for (int w = 0; w < 8; ++w) tot += part[w];
  const float inv = rsqrtf(tot / (float)DIM + EPS_F);

  const int f   = s / HW_GRID;
  const int rem = s - f * HW_GRID;
  const int hh  = rem / W_GRID;
  const int ww  = rem - hh * W_GRID;
  const int t_dim = ROT - 2 * (ROT / 3);  // 22
  const int s1    = ROT / 3;              // 21

  for (int p = tid; p < N_HEADS * ROT; p += 256) {
    const int head = p / ROT;
    const int pr   = p - head * ROT;
    const int frow = (pr < t_dim) ? f : (pr < t_dim + s1 ? hh : ww);
    const float ang = freqs[(size_t)frow * ROT + pr];
    float sn, cs;
    __sincosf(ang, &sn, &cs);
    const int base = head * HEAD_DIM + 2 * pr;
    const size_t idx = (size_t)s * DIM + base;
    const float xr = in[idx]     * inv * g[base];
    const float xi = in[idx + 1] * inv * g[base + 1];
    out[idx]     = (bf16_t)((xr * cs - xi * sn) * mul);
    out[idx + 1] = (bf16_t)((xr * sn + xi * cs) * mul);
  }
}

// ---------------------------------------------------------------------------
// Flash attention, frame-causal. One wave per (16-query tile, head).
// Frame = 448 tokens; 448 % 32 == 0 so 32-key blocks never cross the causal
// boundary -> mask-free, EXEC all-ones around every WMMA.
// Q/K row-major [S][DIM]; V TRANSPOSED [DIM][S] so all fragments are b128s.
// ---------------------------------------------------------------------------
__global__ void attn_flash(const bf16_t* __restrict__ Qm,
                           const bf16_t* __restrict__ Km,
                           const bf16_t* __restrict__ Vt,
                           bf16_t* __restrict__ Om) {
  const int qt   = blockIdx.x;
  const int h    = blockIdx.y;
  const int lane = threadIdx.x & 31;
  const int r    = lane & 15;
  const int half = lane >> 4;
  const int q0   = qt * 16;
  const int kmax = ((q0 / HW_GRID) + 1) * HW_GRID;

  __shared__ __align__(32) bf16_t lds_p[16 * 32];

  // Q fragments (4 k-steps over head_dim=128), contiguous v8bf pairs
  const bf16_t* qrow = Qm + (size_t)(q0 + r) * DIM + h * HEAD_DIM;
  v16bf qa[4];
#pragma unroll
  for (int ks = 0; ks < 4; ++ks) {
    const v8bf lo = *(const v8bf*)(qrow + ks * 32 + half * 16);
    const v8bf hi = *(const v8bf*)(qrow + ks * 32 + 8 + half * 16);
#pragma unroll
    for (int e = 0; e < 8; ++e) { qa[ks][e] = lo[e]; qa[ks][e + 8] = hi[e]; }
  }

  v8f o[8];
#pragma unroll
  for (int j = 0; j < 8; ++j) o[j] = (v8f){};
  float m_i[8], l_i[8];
#pragma unroll
  for (int i = 0; i < 8; ++i) { m_i[i] = -INFINITY; l_i[i] = 0.f; }

  for (int k0 = 0; k0 < kmax; k0 += 32) {
    // ---- S = Q @ K^T : two 16x16 score tiles, 4 dim-steps each ----
    const bf16_t* krow0 = Km + (size_t)(k0 + r)      * DIM + h * HEAD_DIM;
    const bf16_t* krow1 = Km + (size_t)(k0 + 16 + r) * DIM + h * HEAD_DIM;
    v8f s0 = (v8f){}, s1 = (v8f){};
#pragma unroll
    for (int ks = 0; ks < 4; ++ks) {
      const v16bf b0 = *(const v16bf*)(krow0 + ks * 32 + 16 * half);
      const v16bf b1 = *(const v16bf*)(krow1 + ks * 32 + 16 * half);
      s0 = __builtin_amdgcn_wmma_f32_16x16x32_bf16(
          false, qa[ks], false, b0, (short)0, s0, false, false);
      s1 = __builtin_amdgcn_wmma_f32_16x16x32_bf16(
          false, qa[ks], false, b1, (short)0, s1, false, false);
    }

    // ---- online softmax; row m = i + 8*half on lanes (half*16 .. +15) ----
#pragma unroll
    for (int i = 0; i < 8; ++i) {
      float v0 = s0[i], v1 = s1[i];                     // 1/sqrt(d) folded in Q
      float mx = fmaxf(v0, v1);
#pragma unroll
      for (int off = 8; off >= 1; off >>= 1) mx = fmaxf(mx, __shfl_xor(mx, off, 16));
      const float mnew  = fmaxf(m_i[i], mx);
      const float alpha = __expf(m_i[i] - mnew);
      const float p0 = __expf(v0 - mnew);
      const float p1 = __expf(v1 - mnew);
      float ps = p0 + p1;
#pragma unroll
      for (int off = 8; off >= 1; off >>= 1) ps += __shfl_xor(ps, off, 16);
      l_i[i] = l_i[i] * alpha + ps;
      m_i[i] = mnew;
#pragma unroll
      for (int j = 0; j < 8; ++j) o[j][i] *= alpha;
      const int prow = i + 8 * half;
      lds_p[prow * 32 + r]      = (bf16_t)p0;
      lds_p[prow * 32 + 16 + r] = (bf16_t)p1;
    }
    __syncthreads();   // single-wave block: resolves to DS waits

    // ---- P as A-fragment: two ds b128 reads per lane ----
    const v8bf plo = *(const v8bf*)(&lds_p[r * 32 + half * 16]);
    const v8bf phi = *(const v8bf*)(&lds_p[r * 32 + 8 + half * 16]);
    v16bf pa;
#pragma unroll
    for (int e = 0; e < 8; ++e) { pa[e] = plo[e]; pa[e + 8] = phi[e]; }
    __syncthreads();

    // ---- O += P @ V : 8 dim-tiles; V^T rows give contiguous key runs ----
#pragma unroll
    for (int j = 0; j < 8; ++j) {
      const v16bf vb = *(const v16bf*)(
          Vt + (size_t)(h * HEAD_DIM + j * 16 + r) * S_TOK + k0 + 16 * half);
      o[j] = __builtin_amdgcn_wmma_f32_16x16x32_bf16(
          false, pa, false, vb, (short)0, o[j], false, false);
    }
  }

  // ---- normalize, write attn (bf16, [S][DIM] head-major inner) ----
#pragma unroll
  for (int j = 0; j < 8; ++j) {
#pragma unroll
    for (int i = 0; i < 8; ++i) {
      const int row = q0 + i + 8 * half;
      const float val = o[j][i] * (1.0f / l_i[i]);
      Om[(size_t)row * DIM + h * HEAD_DIM + j * 16 + r] = (bf16_t)val;
    }
  }
}

// ---------------------------------------------------------------------------
// Host orchestration
// ---------------------------------------------------------------------------
extern "C" void kernel_launch(void* const* d_in, const int* in_sizes, int n_in,
                              void* d_out, int out_size, void* d_ws, size_t ws_size,
                              hipStream_t stream) {
  (void)in_sizes; (void)n_in; (void)out_size; (void)ws_size;

  const float* x     = (const float*)d_in[0];
  const float* freqs = (const float*)d_in[3];
  const float* Wq    = (const float*)d_in[4];
  const float* bq    = (const float*)d_in[5];
  const float* Wk    = (const float*)d_in[6];
  const float* bk    = (const float*)d_in[7];
  const float* Wv    = (const float*)d_in[8];
  const float* bv    = (const float*)d_in[9];
  const float* Wo    = (const float*)d_in[10];
  const float* bo    = (const float*)d_in[11];
  const float* gq    = (const float*)d_in[12];
  const float* gk    = (const float*)d_in[13];
  float* out = (float*)d_out;

  const size_t SD = (size_t)S_TOK * DIM;
  const size_t DD = (size_t)DIM * DIM;

  char* p = (char*)d_ws;
  bf16_t* xb    = (bf16_t*)p; p += SD * sizeof(bf16_t);
  bf16_t* wqt   = (bf16_t*)p; p += DD * sizeof(bf16_t);   // W^T (N x K)
  bf16_t* wkt   = (bf16_t*)p; p += DD * sizeof(bf16_t);
  bf16_t* wvt   = (bf16_t*)p; p += DD * sizeof(bf16_t);
  bf16_t* wot   = (bf16_t*)p; p += DD * sizeof(bf16_t);
  float*  qf    = (float*)p;  p += SD * sizeof(float);
  float*  kf    = (float*)p;  p += SD * sizeof(float);
  float*  vf    = (float*)p;  p += SD * sizeof(float);
  bf16_t* qb    = (bf16_t*)p; p += SD * sizeof(bf16_t);
  bf16_t* kb    = (bf16_t*)p; p += SD * sizeof(bf16_t);
  bf16_t* vt    = (bf16_t*)p; p += SD * sizeof(bf16_t);   // V^T (DIM x S)
  bf16_t* attnb = (bf16_t*)p; p += SD * sizeof(bf16_t);

  // 1) casts: x row-major; weights cast+transposed for contiguous B-fragments
  cast_f32_bf16<<<1024, 256, 0, stream>>>(x, xb, (int)SD);
  dim3 tb(32, 8);
  dim3 wgrid(DIM / 32, DIM / 32);
  cast_transpose_f32_bf16<<<wgrid, tb, 0, stream>>>(Wq, wqt, DIM, DIM);
  cast_transpose_f32_bf16<<<wgrid, tb, 0, stream>>>(Wk, wkt, DIM, DIM);
  cast_transpose_f32_bf16<<<wgrid, tb, 0, stream>>>(Wv, wvt, DIM, DIM);
  cast_transpose_f32_bf16<<<wgrid, tb, 0, stream>>>(Wo, wot, DIM, DIM);

  // 2) QKV projections (M=4032, N=1536, K=1536); wave slab = 16x64
  dim3 ggrid(DIM / 256, S_TOK / 16);
  gemm_bf16_f32<<<ggrid, 128, 0, stream>>>(xb, wqt, bq, qf, S_TOK, DIM, DIM);
  gemm_bf16_f32<<<ggrid, 128, 0, stream>>>(xb, wkt, bk, kf, S_TOK, DIM, DIM);
  gemm_bf16_f32<<<ggrid, 128, 0, stream>>>(xb, wvt, bv, vf, S_TOK, DIM, DIM);

  // 3) RMSNorm + RoPE (+ 1/sqrt(d) into Q); V cast+transpose to [DIM][S]
  rmsnorm_rope_cast<<<S_TOK, 256, 0, stream>>>(qf, gq, freqs, qb, QK_SCALE);
  rmsnorm_rope_cast<<<S_TOK, 256, 0, stream>>>(kf, gk, freqs, kb, 1.0f);
  dim3 vgrid(DIM / 32, S_TOK / 32);
  cast_transpose_f32_bf16<<<vgrid, tb, 0, stream>>>(vf, vt, S_TOK, DIM);

  // 4) frame-causal flash attention
  dim3 agrid(S_TOK / 16, N_HEADS);
  attn_flash<<<agrid, 32, 0, stream>>>(qb, kb, vt, attnb);

  // 5) output projection -> f32 d_out
  gemm_bf16_f32<<<ggrid, 128, 0, stream>>>(attnb, wot, bo, out, S_TOK, DIM, DIM);
}
